// VanillaPLRNN_60971355734110
// MI455X (gfx1250) — compile-verified, matrix operand order checked by reading, not run on hardware
//
#include <hip/hip_runtime.h>

// ---------------------------------------------------------------------------
// Types for CDNA5 WMMA
// ---------------------------------------------------------------------------
typedef __attribute__((ext_vector_type(2)))  float   v2f;
typedef __attribute__((ext_vector_type(8)))  float   v8f;
typedef __attribute__((ext_vector_type(8)))  __bf16  v8bf;
typedef __attribute__((ext_vector_type(16))) __bf16  v16bf;

static __device__ __forceinline__ v8f wmma_bf16(v16bf a, v16bf b, v8f c) {
  // D(f32 16x16) = A(16x32 bf16) * B(32x16 bf16) + C
  return __builtin_amdgcn_wmma_f32_16x16x32_bf16(false, a, false, b, (short)0, c,
                                                 false, false);
}

static __device__ __forceinline__ v8f wmma_f32(v2f a, v2f b, v8f c) {
  // D(f32 16x16) = A(16x4 f32) * B(4x16 f32) + C
  return __builtin_amdgcn_wmma_f32_16x16x4_f32(false, a, false, b, (short)0, c,
                                               false, false);
}

// Problem constants
#define BSZ   512
#define T1    496
#define T2    95
#define K1    600   // 12 ch * 50 taps   (19 chunks of 32, zero padded to 608)
#define K2    800   // 25 taps * 32 ch   (25 chunks of 32, exact)
#define NCH1C 19
#define NCH2C 25

// ---------------------------------------------------------------------------
// Weight packing into WMMA B-operand layout (bf16, per-lane 16 K-values)
// element j of lane's v16: K_local = (lane<16?0:8) + (j<8 ? j : j+8)
// ---------------------------------------------------------------------------
__global__ void pack_w1_kernel(const float* __restrict__ w, __bf16* __restrict__ wp) {
  int idx = blockIdx.x * 256 + threadIdx.x;
  if (idx >= NCH1C * 2 * 32 * 16) return;
  int j     = idx & 15;
  int lane  = (idx >> 4) & 31;
  int nt    = (idx >> 9) & 1;
  int chunk = idx >> 10;
  int base  = (lane < 16) ? 0 : 8;
  int Kl    = base + (j < 8 ? j : j + 8);
  int Kg    = chunk * 32 + Kl;          // Kg = c*50 + k  (contiguous in conv1_w)
  int o     = nt * 16 + (lane & 15);
  float v   = (Kg < K1) ? w[o * K1 + Kg] : 0.0f;
  wp[idx]   = (__bf16)v;
}

__global__ void pack_w2_kernel(const float* __restrict__ w, __bf16* __restrict__ wp) {
  int idx = blockIdx.x * 256 + threadIdx.x;
  if (idx >= NCH2C * 2 * 32 * 16) return;
  int j     = idx & 15;
  int lane  = (idx >> 4) & 31;
  int nt    = (idx >> 9) & 1;
  int chunk = idx >> 10;
  int base  = (lane < 16) ? 0 : 8;
  int Kl    = base + (j < 8 ? j : j + 8);
  int Kg    = chunk * 32 + Kl;          // Kg = k*32 + f
  int k     = Kg >> 5;
  int f     = Kg & 31;
  int o     = nt * 16 + (lane & 15);
  wp[idx]   = (__bf16)w[o * 800 + f * 25 + k];   // conv2_w[o][f][k]
}

// ---------------------------------------------------------------------------
// conv1 implicit GEMM: y1[b,t,o] = sum_{c,k} x[b,c,10t+k] * w1[o,c,k] + b1[o]
// M = 512*496 rows (64/block), N = 32, K = 600. bf16 WMMA, f32 accumulate.
// Per-block BN partial sums (sum, sumsq per channel) -> part[blk*64 + idx].
// ---------------------------------------------------------------------------
__global__ __launch_bounds__(128) void conv1_wmma_kernel(
    const float* __restrict__ x, const __bf16* __restrict__ w1p,
    const float* __restrict__ bias, __bf16* __restrict__ y1T,
    float* __restrict__ part) {
  __shared__ __bf16 sa[64][40];     // 64 rows x 32 K (padded to 40 -> 16B aligned)
  __shared__ float  wred[256];

  int tid  = threadIdx.x;
  int lane = tid & 31;
  int wv   = tid >> 5;
  int rowBase = blockIdx.x * 64;

  // staging mapping: 2 threads per row, 16 K each
  int rowLocal = tid >> 1;
  int halfo    = (tid & 1) << 4;
  int rowS = rowBase + rowLocal;
  int bS = rowS / T1;
  int tS = rowS - bS * T1;
  const float* xb = x + (size_t)bS * (12 * 5000) + tS * 10;

  v8f acc0 = {}, acc1 = {};
  int mrow = wv * 16 + (lane & 15);
  int c0   = (lane < 16) ? 0 : 8;

  for (int chunk = 0; chunk < NCH1C; ++chunk) {
    __syncthreads();
    int Kb = chunk * 32 + halfo;
    // prefetch next chunk's stream (global_prefetch_b8)
    int Kn = Kb + 32;
    if (Kn < K1) {
      int cn = Kn / 50; int kn = Kn - cn * 50;
      __builtin_prefetch(&xb[cn * 5000 + kn], 0, 1);
    }
#pragma unroll
    for (int i = 0; i < 16; ++i) {
      int Kg = Kb + i;
      float v = 0.0f;
      if (Kg < K1) { int c = Kg / 50; int k = Kg - c * 50; v = xb[c * 5000 + k]; }
      sa[rowLocal][halfo + i] = (__bf16)v;
    }
    __syncthreads();

    v8bf lo = *(const v8bf*)&sa[mrow][c0];
    v8bf hi = *(const v8bf*)&sa[mrow][c0 + 16];
    v16bf a = __builtin_shufflevector(lo, hi, 0,1,2,3,4,5,6,7,8,9,10,11,12,13,14,15);
    const v16bf* wp = (const v16bf*)(w1p + ((size_t)(chunk * 2) * 32 + lane) * 16);
    v16bf b0 = wp[0];
    v16bf b1 = wp[32];
    acc0 = wmma_bf16(a, b0, acc0);
    acc1 = wmma_bf16(a, b1, acc1);
  }

  // epilogue: bias, store bf16 y1T[row][o], BN1 partial stats
  int o0 = lane & 15, o1 = 16 + o0;
  float bb0 = bias[o0], bb1 = bias[o1];
  float s0 = 0, q0 = 0, s1 = 0, q1 = 0;
  int mG = rowBase + wv * 16 + ((lane < 16) ? 0 : 8);
#pragma unroll
  for (int v = 0; v < 8; ++v) {
    float v0 = acc0[v] + bb0;
    float v1 = acc1[v] + bb1;
    s0 += v0; q0 += v0 * v0;
    s1 += v1; q1 += v1 * v1;
    size_t row = (size_t)(mG + v);
    y1T[row * 32 + o0] = (__bf16)v0;
    y1T[row * 32 + o1] = (__bf16)v1;
  }
  s0 += __shfl_xor(s0, 16); q0 += __shfl_xor(q0, 16);
  s1 += __shfl_xor(s1, 16); q1 += __shfl_xor(q1, 16);
  __syncthreads();
  if (lane < 16) {
    wred[wv * 64 +  0 + lane] = s0;
    wred[wv * 64 + 16 + lane] = q0;
    wred[wv * 64 + 32 + lane] = s1;
    wred[wv * 64 + 48 + lane] = q1;
  }
  __syncthreads();
  if (tid < 64) {               // idx 0..31 = sum[ch], 32..63 = sumsq[ch]
    int ch = tid & 31, isq = tid >> 5;
    int slot = ((ch < 16) ? 0 : 2) + isq;
    float v = 0.0f;
#pragma unroll
    for (int w = 0; w < 4; ++w) v += wred[w * 64 + slot * 16 + (ch & 15)];
    part[(size_t)blockIdx.x * 64 + tid] = v;
  }
}

// ---------------------------------------------------------------------------
// Deterministic BN finalize: reduce per-block partials, compute scale/shift
// ---------------------------------------------------------------------------
__global__ void bn_finalize_kernel(const float* __restrict__ part, int nblk,
                                   const float* __restrict__ g,
                                   const float* __restrict__ b,
                                   float* __restrict__ scale,
                                   float* __restrict__ shift, float invCnt) {
  __shared__ float tot[64];
  int j = threadIdx.x;          // 64 threads
  float v = 0.0f;
  for (int i = 0; i < nblk; ++i) v += part[(size_t)i * 64 + j];
  tot[j] = v;
  __syncthreads();
  if (j < 32) {
    float mean = tot[j] * invCnt;
    float var  = tot[32 + j] * invCnt - mean * mean;
    float sc   = g[j] * rsqrtf(var + 1e-5f);
    scale[j] = sc;
    shift[j] = b[j] - mean * sc;
  }
}

// ---------------------------------------------------------------------------
// conv2 implicit GEMM with fused BN1+ReLU on the A-staging path.
// y2[b,t,o] = sum_{f,k} relu(bn1(y1[b,5t+k,f])) * w2[o,f,k] + b2[o]
// K order = k*32 + f so each 32-chunk is one contiguous feature row of y1T.
// ---------------------------------------------------------------------------
__global__ __launch_bounds__(128) void conv2_wmma_kernel(
    const __bf16* __restrict__ y1T, const __bf16* __restrict__ w2p,
    const float* __restrict__ bias, const float* __restrict__ bn1s,
    const float* __restrict__ bn1h, __bf16* __restrict__ y2T,
    float* __restrict__ part) {
  __shared__ __bf16 sa[64][40];
  __shared__ float  wred[256];
  __shared__ float  sbn[64];

  int tid  = threadIdx.x;
  int lane = tid & 31;
  int wv   = tid >> 5;
  int rowBase = blockIdx.x * 64;
  if (tid < 32) { sbn[tid] = bn1s[tid]; sbn[32 + tid] = bn1h[tid]; }

  int rowLocal = tid >> 1;
  int halfo    = (tid & 1) << 4;
  int rowS = rowBase + rowLocal;
  int bS = rowS / T2;
  int tS = rowS - bS * T2;

  v8f acc0 = {}, acc1 = {};
  int mrow = wv * 16 + (lane & 15);
  int c0   = (lane < 16) ? 0 : 8;

  for (int chunk = 0; chunk < NCH2C; ++chunk) {
    __syncthreads();
    int tt = tS * 5 + chunk;                               // k == chunk
    const __bf16* yr = y1T + ((size_t)bS * T1 + tt) * 32 + halfo;
    if (chunk + 1 < NCH2C)                                  // prefetch next row
      __builtin_prefetch(yr + 32, 0, 1);
#pragma unroll
    for (int i = 0; i < 16; ++i) {
      int f = halfo + i;
      float v = (float)yr[i];
      v = fmaxf(v * sbn[f] + sbn[32 + f], 0.0f);           // BN1 + ReLU fused
      sa[rowLocal][halfo + i] = (__bf16)v;
    }
    __syncthreads();

    v8bf lo = *(const v8bf*)&sa[mrow][c0];
    v8bf hi = *(const v8bf*)&sa[mrow][c0 + 16];
    v16bf a = __builtin_shufflevector(lo, hi, 0,1,2,3,4,5,6,7,8,9,10,11,12,13,14,15);
    const v16bf* wp = (const v16bf*)(w2p + ((size_t)(chunk * 2) * 32 + lane) * 16);
    v16bf b0 = wp[0];
    v16bf b1 = wp[32];
    acc0 = wmma_bf16(a, b0, acc0);
    acc1 = wmma_bf16(a, b1, acc1);
  }

  int o0 = lane & 15, o1 = 16 + o0;
  float bb0 = bias[o0], bb1 = bias[o1];
  float s0 = 0, q0 = 0, s1 = 0, q1 = 0;
  int mG = rowBase + wv * 16 + ((lane < 16) ? 0 : 8);
#pragma unroll
  for (int v = 0; v < 8; ++v) {
    float v0 = acc0[v] + bb0;
    float v1 = acc1[v] + bb1;
    s0 += v0; q0 += v0 * v0;
    s1 += v1; q1 += v1 * v1;
    size_t row = (size_t)(mG + v);
    y2T[row * 32 + o0] = (__bf16)v0;
    y2T[row * 32 + o1] = (__bf16)v1;
  }
  s0 += __shfl_xor(s0, 16); q0 += __shfl_xor(q0, 16);
  s1 += __shfl_xor(s1, 16); q1 += __shfl_xor(q1, 16);
  __syncthreads();
  if (lane < 16) {
    wred[wv * 64 +  0 + lane] = s0;
    wred[wv * 64 + 16 + lane] = q0;
    wred[wv * 64 + 32 + lane] = s1;
    wred[wv * 64 + 48 + lane] = q1;
  }
  __syncthreads();
  if (tid < 64) {
    int ch = tid & 31, isq = tid >> 5;
    int slot = ((ch < 16) ? 0 : 2) + isq;
    float v = 0.0f;
#pragma unroll
    for (int w = 0; w < 4; ++w) v += wred[w * 64 + slot * 16 + (ch & 15)];
    part[(size_t)blockIdx.x * 64 + tid] = v;
  }
}

// ---------------------------------------------------------------------------
// BN2 + ReLU + projection (32x32) + LayerNorm.  One wave per (b,t) row.
// lane f supplies feat via shuffle broadcast; lane l owns output l.
// ---------------------------------------------------------------------------
__global__ __launch_bounds__(256) void proj_ln_kernel(
    const __bf16* __restrict__ y2T, const float* __restrict__ bn2s,
    const float* __restrict__ bn2h, const float* __restrict__ proj_w,
    const float* __restrict__ proj_b, const float* __restrict__ ln_g,
    const float* __restrict__ ln_b, float* __restrict__ xt) {
  __shared__ float spw[32 * 33];
  int tid = threadIdx.x;
  for (int i = tid; i < 32 * 32; i += 256)
    spw[(i >> 5) * 33 + (i & 31)] = proj_w[i];
  __syncthreads();

  int lane = tid & 31;
  int wv   = tid >> 5;
  int gRow = blockIdx.x * 8 + wv;     // exact: 48640 rows / 8 waves
  float feat = (float)y2T[(size_t)gRow * 32 + lane];
  feat = fmaxf(feat * bn2s[lane] + bn2h[lane], 0.0f);

  float acc = proj_b[lane];
#pragma unroll
  for (int f = 0; f < 32; ++f) {
    float v = __shfl(feat, f);
    acc += v * spw[lane * 33 + f];
  }
  // LayerNorm across the 32 lanes
  float m = acc;
#pragma unroll
  for (int off = 16; off > 0; off >>= 1) m += __shfl_xor(m, off);
  m *= (1.0f / 32.0f);
  float d = acc - m;
  float vv = d * d;
#pragma unroll
  for (int off = 16; off > 0; off >>= 1) vv += __shfl_xor(vv, off);
  vv *= (1.0f / 32.0f);
  xt[(size_t)gRow * 32 + lane] = d * rsqrtf(vv + 1e-5f) * ln_g[lane] + ln_b[lane];
}

// ---------------------------------------------------------------------------
// PLRNN recurrence with V_WMMA_F32_16X16X4_F32.
// One wave handles 16 batch rows.  State h kept as two 16x16 D-layout tiles
// (M = latent, N = batch).  B-operands rebuilt from D-layout via 4 shuffles
// per K-chunk of 4.  Per step: 16 WMMA (h A^T) + 32 WMMA (h W) + 32 WMMA
// (relu @ W^T), all accumulator-chained; mean-pool and out-proj fused.
// x_t tile staged each step with GLOBAL_LOAD_ASYNC_TO_LDS_B128 (ASYNCcnt).
// ---------------------------------------------------------------------------
#define SXP 36   // sX row stride in floats: 144B = 9*16B -> every b128 dest aligned
__global__ __launch_bounds__(32) void plrnn_scan_kernel(
    const float* __restrict__ xt, const float* __restrict__ Amat,
    const float* __restrict__ Wmat, const float* __restrict__ h_bias,
    const float* __restrict__ out_w, const float* __restrict__ out_b,
    float* __restrict__ out) {
  __shared__ float sA[32 * 34];    // A[l2][l]
  __shared__ float sWT[64 * 34];   // W^T[j][l]
  __shared__ float sW[32 * 66];    // W[l2][j]
  __shared__ float sOW[32 * 34];   // out_w[l2][l]
  __shared__ float sX[16 * SXP];   // x_t staging [batch][latent]

  int lane = threadIdx.x;
  int b0 = blockIdx.x * 16;
  for (int i = lane; i < 32 * 32; i += 32) {
    int r = i >> 5, c = i & 31;
    sA[r * 34 + c]  = Amat[i];
    sOW[r * 34 + c] = out_w[i];
  }
  for (int i = lane; i < 64 * 32; i += 32) {
    int j = i >> 5, l = i & 31;
    sWT[j * 34 + l] = Wmat[l * 64 + j];
  }
  for (int i = lane; i < 32 * 64; i += 32) {
    int l2 = i >> 6, j = i & 63;
    sW[l2 * 66 + j] = Wmat[i];
  }
  __syncthreads();

  int  nn   = lane & 15;
  bool lo16 = lane < 16;
  int  msel = lo16 ? 0 : 8;

  v8f h0 = {}, h1 = {}, ac0 = {}, ac1 = {};
  v8f hbT0, hbT1, hbT2, hbT3;
#pragma unroll
  for (int r = 0; r < 8; ++r) {
    hbT0[r] = h_bias[ 0 + r + msel];
    hbT1[r] = h_bias[16 + r + msel];
    hbT2[r] = h_bias[32 + r + msel];
    hbT3[r] = h_bias[48 + r + msel];
  }

  for (int t = 0; t < T2; ++t) {
    // ---- stage x_t tile (16 rows x 128B) via async global->LDS DMA -------
    // 4 x b128: each lane moves 16B; chunk id = q*32+lane -> (row, 16B seg)
#pragma unroll
    for (int q = 0; q < 4; ++q) {
      int cid = q * 32 + lane;
      int row = cid >> 3;
      int seg = cid & 7;
      const float* g = xt + (((size_t)(b0 + row) * T2 + t) * 32 + seg * 4);
      unsigned lds_addr = (unsigned)(uintptr_t)&sX[row * SXP + seg * 4];
      asm volatile("global_load_async_to_lds_b128 %0, %1, off"
                   :: "v"(lds_addr), "v"(g) : "memory");
    }
    asm volatile("s_wait_asynccnt 0x0" ::: "memory");
    __syncthreads();

    v8f xv0, xv1;
#pragma unroll
    for (int r = 0; r < 8; ++r) {
      xv0[r] = sX[nn * SXP + r + msel];
      xv1[r] = sX[nn * SXP + 16 + r + msel];
    }

    // B-operand chunks (K over latent) from D-layout h
    v2f Bc[8];
#pragma unroll
    for (int kk = 0; kk < 8; ++kk) {
      const int kbase = kk * 4;
      const v8f Ht = (kbase < 16) ? h0 : h1;
      const int r0 = kbase & 7;
      int srcLane = nn + ((kbase & 8) ? 16 : 0);
      float e0 = __shfl(Ht[r0 + 0], srcLane);
      float e1 = __shfl(Ht[r0 + 1], srcLane);
      float e2 = __shfl(Ht[r0 + 2], srcLane);
      float e3 = __shfl(Ht[r0 + 3], srcLane);
      Bc[kk][0] = lo16 ? e0 : e2;
      Bc[kk][1] = lo16 ? e1 : e3;
    }

    // linear part: L = x_t + h @ A^T  (C initialized with x)
    v8f L0 = xv0, L1 = xv1;
#pragma unroll
    for (int kk = 0; kk < 8; ++kk) {
      int col = kk * 4 + (lo16 ? 0 : 2);
      v2f a0 = *(const v2f*)&sA[( 0 + nn) * 34 + col];
      v2f a1 = *(const v2f*)&sA[(16 + nn) * 34 + col];
      L0 = wmma_f32(a0, Bc[kk], L0);
      L1 = wmma_f32(a1, Bc[kk], L1);
    }

    // hidden: Hd = relu(h @ W + h_bias), 64 rows = 4 M-tiles
    v8f H0 = hbT0, H1 = hbT1, H2 = hbT2, H3 = hbT3;
#pragma unroll
    for (int kk = 0; kk < 8; ++kk) {
      int col = kk * 4 + (lo16 ? 0 : 2);
      v2f a0 = *(const v2f*)&sWT[( 0 + nn) * 34 + col];
      v2f a1 = *(const v2f*)&sWT[(16 + nn) * 34 + col];
      v2f a2 = *(const v2f*)&sWT[(32 + nn) * 34 + col];
      v2f a3 = *(const v2f*)&sWT[(48 + nn) * 34 + col];
      H0 = wmma_f32(a0, Bc[kk], H0);
      H1 = wmma_f32(a1, Bc[kk], H1);
      H2 = wmma_f32(a2, Bc[kk], H2);
      H3 = wmma_f32(a3, Bc[kk], H3);
    }
#pragma unroll
    for (int r = 0; r < 8; ++r) {
      H0[r] = fmaxf(H0[r], 0.0f); H1[r] = fmaxf(H1[r], 0.0f);
      H2[r] = fmaxf(H2[r], 0.0f); H3[r] = fmaxf(H3[r], 0.0f);
    }

    // nonlinear: h_new = L + Hd @ W^T  (chained on L accumulators)
#pragma unroll
    for (int kk = 0; kk < 16; ++kk) {
      const int kbase = kk * 4;
      const v8f Ht = (kbase < 16) ? H0 : (kbase < 32) ? H1 : (kbase < 48) ? H2 : H3;
      const int r0 = kbase & 7;
      int srcLane = nn + ((kbase & 8) ? 16 : 0);
      float e0 = __shfl(Ht[r0 + 0], srcLane);
      float e1 = __shfl(Ht[r0 + 1], srcLane);
      float e2 = __shfl(Ht[r0 + 2], srcLane);
      float e3 = __shfl(Ht[r0 + 3], srcLane);
      v2f Bh;
      Bh[0] = lo16 ? e0 : e2;
      Bh[1] = lo16 ? e1 : e3;
      int col = kbase + (lo16 ? 0 : 2);
      v2f a0 = *(const v2f*)&sW[( 0 + nn) * 66 + col];
      v2f a1 = *(const v2f*)&sW[(16 + nn) * 66 + col];
      L0 = wmma_f32(a0, Bh, L0);
      L1 = wmma_f32(a1, Bh, L1);
    }
    h0 = L0; h1 = L1;
#pragma unroll
    for (int r = 0; r < 8; ++r) { ac0[r] += h0[r]; ac1[r] += h1[r]; }
    __syncthreads();
  }

  // mean pool + output projection
  const float inv = 1.0f / (float)T2;
#pragma unroll
  for (int r = 0; r < 8; ++r) { ac0[r] *= inv; ac1[r] *= inv; }

  v8f O0, O1;
#pragma unroll
  for (int r = 0; r < 8; ++r) {
    O0[r] = out_b[ 0 + r + msel];
    O1[r] = out_b[16 + r + msel];
  }
#pragma unroll
  for (int kk = 0; kk < 8; ++kk) {
    const int kbase = kk * 4;
    const v8f Pt = (kbase < 16) ? ac0 : ac1;
    const int r0 = kbase & 7;
    int srcLane = nn + ((kbase & 8) ? 16 : 0);
    float e0 = __shfl(Pt[r0 + 0], srcLane);
    float e1 = __shfl(Pt[r0 + 1], srcLane);
    float e2 = __shfl(Pt[r0 + 2], srcLane);
    float e3 = __shfl(Pt[r0 + 3], srcLane);
    v2f Bp;
    Bp[0] = lo16 ? e0 : e2;
    Bp[1] = lo16 ? e1 : e3;
    int col = kbase + (lo16 ? 0 : 2);
    v2f a0 = *(const v2f*)&sOW[( 0 + nn) * 34 + col];
    v2f a1 = *(const v2f*)&sOW[(16 + nn) * 34 + col];
    O0 = wmma_f32(a0, Bp, O0);
    O1 = wmma_f32(a1, Bp, O1);
  }
#pragma unroll
  for (int r = 0; r < 8; ++r) {
    out[(size_t)(b0 + nn) * 32 +  0 + r + msel] = O0[r];
    out[(size_t)(b0 + nn) * 32 + 16 + r + msel] = O1[r];
  }
}

// ---------------------------------------------------------------------------
// Host launcher
// ---------------------------------------------------------------------------
extern "C" void kernel_launch(void* const* d_in, const int* in_sizes, int n_in,
                              void* d_out, int out_size, void* d_ws, size_t ws_size,
                              hipStream_t stream) {
  (void)in_sizes; (void)n_in; (void)out_size; (void)ws_size;
  const float* x       = (const float*)d_in[0];
  const float* conv1_w = (const float*)d_in[1];
  const float* conv1_b = (const float*)d_in[2];
  const float* bn1_g   = (const float*)d_in[3];
  const float* bn1_b   = (const float*)d_in[4];
  const float* conv2_w = (const float*)d_in[5];
  const float* conv2_b = (const float*)d_in[6];
  const float* bn2_g   = (const float*)d_in[7];
  const float* bn2_b   = (const float*)d_in[8];
  const float* proj_w  = (const float*)d_in[9];
  const float* proj_b  = (const float*)d_in[10];
  const float* ln_g    = (const float*)d_in[11];
  const float* ln_b    = (const float*)d_in[12];
  const float* Amat    = (const float*)d_in[13];
  const float* Wmat    = (const float*)d_in[14];
  const float* h_bias  = (const float*)d_in[15];
  const float* out_w   = (const float*)d_in[16];
  const float* out_b   = (const float*)d_in[17];

  char* ws = (char*)d_ws;
  __bf16* y1T  = (__bf16*)(ws + 0);            // 512*496*32 bf16 = 16,252,928 B
  __bf16* y2T  = (__bf16*)(ws + 16252928);     // 512*95*32 bf16  =  3,112,960 B
  float*  xt   = (float*) (ws + 19365888);     // 512*95*32 f32   =  6,225,920 B
  __bf16* w1p  = (__bf16*)(ws + 25591808);     // 19*2*32*16 bf16 =     38,912 B
  __bf16* w2p  = (__bf16*)(ws + 25630720);     // 25*2*32*16 bf16 =     51,200 B
  float*  part1 = (float*)(ws + 25681920);     // 3968*64 f32     =  1,015,808 B
  float*  part2 = (float*)(ws + 26697728);     // 760*64 f32      =    194,560 B
  float*  bnp   = (float*)(ws + 26892288);     // bn1s/bn1h/bn2s/bn2h, 32 each

  const int NBLK1 = (BSZ * T1) / 64;   // 3968
  const int NBLK2 = (BSZ * T2) / 64;   // 760

  pack_w1_kernel<<<(NCH1C * 1024 + 255) / 256, 256, 0, stream>>>(conv1_w, w1p);
  pack_w2_kernel<<<(NCH2C * 1024 + 255) / 256, 256, 0, stream>>>(conv2_w, w2p);

  conv1_wmma_kernel<<<NBLK1, 128, 0, stream>>>(x, w1p, conv1_b, y1T, part1);
  bn_finalize_kernel<<<1, 64, 0, stream>>>(part1, NBLK1, bn1_g, bn1_b,
                                           bnp + 0, bnp + 32,
                                           1.0f / (float)(BSZ * T1));
  conv2_wmma_kernel<<<NBLK2, 128, 0, stream>>>(y1T, w2p, conv2_b, bnp + 0, bnp + 32,
                                               y2T, part2);
  bn_finalize_kernel<<<1, 64, 0, stream>>>(part2, NBLK2, bn2_g, bn2_b,
                                           bnp + 64, bnp + 96,
                                           1.0f / (float)(BSZ * T2));
  proj_ln_kernel<<<(BSZ * T2) / 8, 256, 0, stream>>>(y2T, bnp + 64, bnp + 96,
                                                     proj_w, proj_b, ln_g, ln_b, xt);
  plrnn_scan_kernel<<<BSZ / 16, 32, 0, stream>>>(xt, Amat, Wmat, h_bias,
                                                 out_w, out_b, (float*)d_out);
}
